// HelixNetHybridAttention_6760278524024
// MI455X (gfx1250) — matrix operand order repaired and checked by reading, not compiled, not run on hardware
//
#include <hip/hip_runtime.h>
#include <hip/hip_bf16.h>

// ---------------- types / helpers ----------------
typedef __bf16 bf16x16 __attribute__((ext_vector_type(16)));
typedef float  f32x8   __attribute__((ext_vector_type(8)));

#define WMMA_BF16(a, b, c) \
  __builtin_amdgcn_wmma_f32_16x16x32_bf16(false, (a), false, (b), (short)0, (c), false, false)

__device__ __forceinline__ __bf16 us2bf(unsigned short u) {
  union { unsigned short s; __bf16 b; } x; x.s = u; return x.b;
}
__device__ __forceinline__ unsigned short f2us(float f) {
  union { float f; unsigned int u; } x; x.f = f;
  unsigned int u = x.u;
  u += 0x7FFFu + ((u >> 16) & 1u);       // round-to-nearest-even
  return (unsigned short)(u >> 16);
}
__device__ __forceinline__ float us2f(unsigned short u) {
  union { unsigned int u; float f; } x; x.u = ((unsigned int)u) << 16; return x.f;
}
__device__ __forceinline__ __bf16 f2bf(float f) { return us2bf(f2us(f)); }

// ---------------- Tensor Data Mover (CDNA5 TDM) ----------------
#if defined(__has_builtin)
#if __has_builtin(__builtin_amdgcn_tensor_load_to_lds)
#define HAVE_TDM 1
#endif
#endif
#ifndef HAVE_TDM
#define HAVE_TDM 0
#endif

__device__ __forceinline__ void wait_tensorcnt0() {
#if defined(__has_builtin) && __has_builtin(__builtin_amdgcn_s_wait_tensorcnt)
  __builtin_amdgcn_s_wait_tensorcnt(0);
#else
  asm volatile("s_wait_tensorcnt 0x0" ::: "memory");
#endif
}

#if HAVE_TDM
typedef unsigned int tdm_u32x4 __attribute__((ext_vector_type(4)));
typedef int          tdm_i32x8 __attribute__((ext_vector_type(8)));
typedef int          tdm_i32x4 __attribute__((ext_vector_type(4)));

// D# descriptor per CDNA5 ISA sec.8.
// 2D tile (tile_d1>0, row stride stride0_elems) or 1D copy (tile_d1==0).
// tensor dims == tile dims -> no OOB clamping. data size: ds_log 1 => 2-byte elems.
__device__ __forceinline__ void tdm_load_to_lds(unsigned lds_byte_addr, const void* gptr,
                                                unsigned tile_d0, unsigned tile_d1,
                                                unsigned long long stride0_elems,
                                                unsigned ds_log) {
  unsigned long long ga = (unsigned long long)(size_t)gptr;
  tdm_u32x4 g0;
  g0[0] = 1u;                                               // count=1 (valid user D#)
  g0[1] = lds_byte_addr;                                    // LDS byte address
  g0[2] = (unsigned)ga;                                     // global_addr[31:0]
  g0[3] = (unsigned)((ga >> 32) & 0x1FFFFFFu) | (2u << 30); // addr[56:32] | type=2
  tdm_i32x8 g1;
  g1[0] = (int)(ds_log << 16);                                              // data_size
  g1[1] = (int)((tile_d0 & 0xFFFFu) << 16);                                 // tensor_dim0 lo
  g1[2] = (int)(((tile_d0 >> 16) & 0xFFFFu) | ((tile_d1 & 0xFFFFu) << 16)); // dim0 hi | dim1 lo
  g1[3] = (int)(((tile_d1 >> 16) & 0xFFFFu) | ((tile_d0 & 0xFFFFu) << 16)); // dim1 hi | tile_dim0
  g1[4] = (int)(tile_d1 & 0xFFFFu);                                         // tile_dim1 (tile_dim2=0)
  g1[5] = (int)(unsigned)(stride0_elems & 0xFFFFFFFFull);                   // dim0_stride lo32
  g1[6] = (int)(unsigned)((stride0_elems >> 32) & 0xFFFFull);               // dim0_stride hi16
  g1[7] = 0;
  tdm_i32x4 z4 = {0, 0, 0, 0};
#if __clang_major__ >= 23
  tdm_i32x8 z8 = {0, 0, 0, 0, 0, 0, 0, 0};
  __builtin_amdgcn_tensor_load_to_lds(g0, g1, z4, z4, z8, 0);
#else
  __builtin_amdgcn_tensor_load_to_lds(g0, g1, z4, z4, 0);
#endif
}
#endif  // HAVE_TDM

// ---------------- fp32 -> bf16 convert ----------------
__global__ __launch_bounds__(256) void cvt_bf16_kernel(const float* __restrict__ src,
                                                       unsigned short* __restrict__ dst, int n) {
  int i = blockIdx.x * 256 + threadIdx.x;
  if (i < n) dst[i] = f2us(src[i]);
}

// ---------------- generic WMMA GEMM: C[M,N] fp32 = A_bf16[M,K] @ B_bf16[K,N] ----------------
// BM=128 BN=128 BK=32, 256 threads = 8 waves (4 x 2), each wave 32x64 (2x4 wmma tiles)
// A tile staged via TDM (global->LDS DMA, TENSORcnt); B tile transposed via VGPRs (overlaps TDM).
__global__ __launch_bounds__(256) void gemm_bf16_kernel(const unsigned short* __restrict__ A,
                                                        const unsigned short* __restrict__ Bm,
                                                        float* __restrict__ C,
                                                        int M, int N, int K) {
  extern __shared__ char gsm[];
  unsigned short* As = (unsigned short*)gsm;            // row-major [128][32], LDS bytes [0,8192)
  unsigned short* Bs = (unsigned short*)(gsm + 8192);   // column-major Bs[n*32+k], [8192,16384)
  const int tid  = threadIdx.x;
  const int lane = tid & 31, wave = tid >> 5;
  const int wm = wave & 3, wn = wave >> 2;
  const int bm = blockIdx.y * 128, bn = blockIdx.x * 128;
  const int lane15 = lane & 15;
  const int k0l  = (lane >> 4) * 8;
  const int kOff = (lane >> 4) * 16;
  const int mr8  = (lane >> 4) * 8;

  f32x8 acc[2][4];
  for (int i = 0; i < 2; i++)
    for (int j = 0; j < 4; j++)
      for (int e = 0; e < 8; e++) acc[i][j][e] = 0.f;

  for (int k0 = 0; k0 < K; k0 += 32) {
#if HAVE_TDM
    if (wave == 0)   // one TDM op: 128 rows x 32 bf16, row stride K -> packed LDS
      tdm_load_to_lds(0u, (const char*)A + ((size_t)bm * K + k0) * 2, 32u, 128u,
                      (unsigned long long)(unsigned)K, 1u);
#else
    {
      const unsigned int* Ag = (const unsigned int*)A;
      #pragma unroll
      for (int i = 0; i < 8; ++i) {                 // 2048 uints = 128x32 bf16
        int idx = tid + i * 256;
        int r = idx >> 4, cu = idx & 15;
        ((unsigned int*)As)[r * 16 + cu] = Ag[(size_t)(bm + r) * (K >> 1) + (k0 >> 1) + cu];
      }
    }
#endif
    {
      const unsigned int* Bg = (const unsigned int*)Bm;
      #pragma unroll
      for (int i = 0; i < 8; ++i) {                 // 2048 uints = 32x128 bf16 -> col-major
        int idx = tid + i * 256;
        int r = idx >> 6, nu = idx & 63;
        unsigned int w = Bg[(size_t)(k0 + r) * (N >> 1) + (bn >> 1) + nu];
        Bs[(2 * nu) * 32 + r]     = (unsigned short)(w & 0xFFFFu);
        Bs[(2 * nu + 1) * 32 + r] = (unsigned short)(w >> 16);
      }
    }
    if (k0 + 32 < K) {   // speculative prefetch of next tiles (global_prefetch_b8)
      __builtin_prefetch((const char*)A + ((size_t)(bm + (tid >> 1)) * K + k0 + 32) * 2, 0, 0);
      __builtin_prefetch((const char*)Bm + ((size_t)(k0 + 32 + (tid >> 3)) * N + bn) * 2, 0, 0);
    }
#if HAVE_TDM
    if (wave == 0) wait_tensorcnt0();
#endif
    __syncthreads();

    bf16x16 afr[2];
    #pragma unroll
    for (int im = 0; im < 2; ++im) {
      int row = wm * 32 + im * 16 + lane15;
      #pragma unroll
      for (int i = 0; i < 8; i++) {
        afr[im][i]     = us2bf(As[row * 32 + k0l + i]);
        afr[im][8 + i] = us2bf(As[row * 32 + k0l + 16 + i]);
      }
    }
    #pragma unroll
    for (int in = 0; in < 4; ++in) {
      int n = wn * 64 + in * 16 + lane15;
      bf16x16 bfr;
      #pragma unroll
      for (int i = 0; i < 16; i++) bfr[i] = us2bf(Bs[n * 32 + k0l + (i & 8) + i]);
      #pragma unroll
      for (int im = 0; im < 2; ++im) acc[im][in] = WMMA_BF16(afr[im], bfr, acc[im][in]);
    }
    __syncthreads();
  }

  #pragma unroll
  for (int im = 0; im < 2; ++im)
    #pragma unroll
    for (int in = 0; in < 4; ++in) {
      int col = bn + wn * 64 + in * 16 + lane15;
      #pragma unroll
      for (int r = 0; r < 8; r++) {
        int row = bm + wm * 32 + im * 16 + mr8 + r;
        C[(size_t)row * N + col] = acc[im][in][r];
      }
    }
}

// ---------------- beta/decay: sigmoid(h @ W[2048,4] (+bias)) ----------------
__global__ __launch_bounds__(256) void beta_decay_kernel(const float* __restrict__ hsrc,
                                                         const float* __restrict__ Wb,
                                                         const float* __restrict__ Wd,
                                                         const float* __restrict__ bdec,
                                                         float* __restrict__ betaf,
                                                         float* __restrict__ decayf) {
  __shared__ float red[8][8];
  int row = blockIdx.x, tid = threadIdx.x, lane = tid & 31, wave = tid >> 5;
  float ab[4] = {0, 0, 0, 0}, ad[4] = {0, 0, 0, 0};
  for (int k = tid; k < 2048; k += 256) {
    float hv = hsrc[(size_t)row * 2048 + k];
    #pragma unroll
    for (int j = 0; j < 4; j++) { ab[j] += hv * Wb[k * 4 + j]; ad[j] += hv * Wd[k * 4 + j]; }
  }
  #pragma unroll
  for (int j = 0; j < 4; j++)
    for (int m = 1; m < 32; m <<= 1) {
      ab[j] += __shfl_xor(ab[j], m, 32);
      ad[j] += __shfl_xor(ad[j], m, 32);
    }
  if (lane == 0)
    for (int j = 0; j < 4; j++) { red[wave][j] = ab[j]; red[wave][4 + j] = ad[j]; }
  __syncthreads();
  if (tid < 4) {
    float sb = 0.f, sd = 0.f;
    for (int w = 0; w < 8; w++) { sb += red[w][tid]; sd += red[w][4 + tid]; }
    betaf[(size_t)row * 4 + tid]  = 1.f / (1.f + __expf(-sb));
    decayf[(size_t)row * 4 + tid] = 1.f / (1.f + __expf(-(sd + bdec[tid])));
  }
}

// ---------------- RoPE + relayout to [B,HA,S,DH] bf16 ----------------
__global__ __launch_bounds__(128) void rope_kernel(const float* __restrict__ qf,
                                                   const float* __restrict__ kf,
                                                   const float* __restrict__ vf,
                                                   unsigned short* __restrict__ qr,
                                                   unsigned short* __restrict__ kr,
                                                   unsigned short* __restrict__ vr) {
  int d = threadIdx.x;                 // 0..127
  int h = blockIdx.y;                  // 0..15
  int row = blockIdx.x;                // b*2048+s
  int s = row & 2047, b = row >> 11;
  size_t gsrc = (size_t)row * 2048 + h * 128 + d;
  float q = qf[gsrc], k = kf[gsrc], v = vf[gsrc];
  int d2 = d & 63;
  float inv = __expf(-(float)(2 * d2) * (1.f / 128.f) * 9.210340372f);  // ln(10000)
  float ang = (float)s * inv;
  float cs = cosf(ang), sn = sinf(ang);
  size_t part = (size_t)row * 2048 + h * 128 + ((d + 64) & 127);
  float qb = qf[part], kb = kf[part];
  float qrot = (d < 64) ? -qb : qb;
  float krot = (d < 64) ? -kb : kb;
  size_t gdst = (((size_t)(b * 16 + h)) * 2048 + s) * 128 + d;
  qr[gdst] = f2us(q * cs + qrot * sn);
  kr[gdst] = f2us(k * cs + krot * sn);
  vr[gdst] = f2us(v);
}

// ---------------- flash attention (window 1024, causal), writes bf16 into cat[:, 0:2048] ------
// LDS layout (dynamic): Qs @0 (16KB), Ks @16384 (16KB), Vs @32768 (16KB), then Ss/Ps/stats.
// Q/K/V tiles are contiguous 16KB blocks -> 1D TDM copies.
__global__ __launch_bounds__(256) void attn_kernel(const unsigned short* __restrict__ qr,
                                                   const unsigned short* __restrict__ kr,
                                                   const unsigned short* __restrict__ vr,
                                                   unsigned short* __restrict__ cat) {
  extern __shared__ char smem[];
  unsigned short* Qs = (unsigned short*)smem;           // 64x128 bf16
  unsigned short* Ks = Qs + 64 * 128;                   // 64x128
  unsigned short* Vs = Ks + 64 * 128;                   // 64x128
  float* Ss          = (float*)(Vs + 64 * 128);         // 64x64 fp32
  unsigned short* Ps = (unsigned short*)(Ss + 64 * 64); // 64x64 bf16
  float* mrow = (float*)(Ps + 64 * 64);
  float* lrow = mrow + 64;
  float* arow = lrow + 64;

  const int tid = threadIdx.x, lane = tid & 31, wave = tid >> 5;
  const int wm = wave & 3, wn = wave >> 2;
  const int lane15 = lane & 15, k0l = (lane >> 4) * 8, kOff = (lane >> 4) * 16, mr8 = (lane >> 4) * 8;
  const int bh = blockIdx.x;                            // b*16 + h
  const int qb0 = blockIdx.y * 64;
  const size_t base = (size_t)bh * 2048 * 128;

#if HAVE_TDM
  if (wave == 0)
    tdm_load_to_lds(0u, qr + base + (size_t)qb0 * 128, 8192u, 0u, 0ull, 1u);
#else
  {
    const unsigned int* Qg = (const unsigned int*)(qr + base + (size_t)qb0 * 128);
    unsigned int* Qsu = (unsigned int*)Qs;
    #pragma unroll
    for (int i = 0; i < 16; i++) Qsu[tid + i * 256] = Qg[tid + i * 256];
  }
#endif
  if (tid < 64) { mrow[tid] = -1e30f; lrow[tid] = 0.f; arow[tid] = 0.f; }
#if HAVE_TDM
  if (wave == 0) wait_tensorcnt0();
#endif
  __syncthreads();

  f32x8 oacc[4];
  for (int j = 0; j < 4; j++) for (int e = 0; e < 8; e++) oacc[j][e] = 0.f;

  int kstart = qb0 - 1024; if (kstart < 0) kstart = 0;
  for (int kb0 = kstart; kb0 <= qb0; kb0 += 64) {
#if HAVE_TDM
    if (wave == 0) {
      tdm_load_to_lds(16384u, kr + base + (size_t)kb0 * 128, 8192u, 0u, 0ull, 1u);
      tdm_load_to_lds(32768u, vr + base + (size_t)kb0 * 128, 8192u, 0u, 0ull, 1u);
      wait_tensorcnt0();
    }
#else
    {
      const unsigned int* Kg = (const unsigned int*)(kr + base + (size_t)kb0 * 128);
      const unsigned int* Vg = (const unsigned int*)(vr + base + (size_t)kb0 * 128);
      unsigned int* Ksu = (unsigned int*)Ks;
      unsigned int* Vsu = (unsigned int*)Vs;
      #pragma unroll
      for (int i = 0; i < 16; i++) { Ksu[tid + i * 256] = Kg[tid + i * 256];
                                     Vsu[tid + i * 256] = Vg[tid + i * 256]; }
    }
#endif
    __syncthreads();

    // ---- scores 64x64: wave -> rows 16*wm, cols 32*wn ----
    f32x8 sacc[2];
    for (int j = 0; j < 2; j++) for (int e = 0; e < 8; e++) sacc[j][e] = 0.f;
    #pragma unroll
    for (int kk = 0; kk < 128; kk += 32) {
      bf16x16 af;
      int row = wm * 16 + lane15;
      #pragma unroll
      for (int i = 0; i < 8; i++) {
        af[i]     = us2bf(Qs[row * 128 + kk + k0l + i]);
        af[8 + i] = us2bf(Qs[row * 128 + kk + k0l + 16 + i]);
      }
      #pragma unroll
      for (int in = 0; in < 2; in++) {
        int n = wn * 32 + in * 16 + lane15;
        bf16x16 bfv;
        #pragma unroll
        for (int i = 0; i < 16; i++) bfv[i] = us2bf(Ks[n * 128 + kk + k0l + (i & 8) + i]);
        sacc[in] = WMMA_BF16(af, bfv, sacc[in]);
      }
    }
    #pragma unroll
    for (int in = 0; in < 2; in++) {
      int col = wn * 32 + in * 16 + lane15;
      int kAbs = kb0 + col;
      #pragma unroll
      for (int r = 0; r < 8; r++) {
        int row = wm * 16 + mr8 + r;
        int qAbs = qb0 + row;
        float vv = sacc[in][r] * 0.08838834764831845f;     // 1/sqrt(128)
        bool ok = (kAbs <= qAbs) && (qAbs - kAbs <= 1024);
        Ss[row * 64 + col] = ok ? vv : -1e30f;
      }
    }
    __syncthreads();

    // ---- online softmax: 4 lanes per row, 16 cols each ----
    {
      int row = tid >> 2, seg = tid & 3;
      float vals[16];
      float mloc = -1e30f;
      #pragma unroll
      for (int i = 0; i < 16; i++) { vals[i] = Ss[row * 64 + seg * 16 + i]; mloc = fmaxf(mloc, vals[i]); }
      mloc = fmaxf(mloc, __shfl_xor(mloc, 1, 4));
      mloc = fmaxf(mloc, __shfl_xor(mloc, 2, 4));
      float mold = mrow[row];
      float mnew = fmaxf(mold, mloc);
      float ssum = 0.f;
      #pragma unroll
      for (int i = 0; i < 16; i++) {
        float p = __expf(vals[i] - mnew);
        ssum += p;
        Ps[row * 64 + seg * 16 + i] = f2us(p);
      }
      ssum += __shfl_xor(ssum, 1, 4);
      ssum += __shfl_xor(ssum, 2, 4);
      float alpha = __expf(mold - mnew);
      if (seg == 0) { lrow[row] = lrow[row] * alpha + ssum; mrow[row] = mnew; arow[row] = alpha; }
    }
    __syncthreads();

    // ---- rescale O, then O += P @ V ----
    {
      float al[8];
      #pragma unroll
      for (int r = 0; r < 8; r++) al[r] = arow[wm * 16 + mr8 + r];
      #pragma unroll
      for (int j = 0; j < 4; j++)
        #pragma unroll
        for (int r = 0; r < 8; r++) oacc[j][r] *= al[r];
    }
    #pragma unroll
    for (int kk = 0; kk < 64; kk += 32) {
      bf16x16 af;
      int row = wm * 16 + lane15;
      #pragma unroll
      for (int i = 0; i < 8; i++) {
        af[i]     = us2bf(Ps[row * 64 + kk + k0l + i]);
        af[8 + i] = us2bf(Ps[row * 64 + kk + k0l + 16 + i]);
      }
      #pragma unroll
      for (int in = 0; in < 4; in++) {
        int n = wn * 64 + in * 16 + lane15;
        bf16x16 bfv;
        #pragma unroll
        for (int i = 0; i < 16; i++) bfv[i] = us2bf(Vs[(kk + k0l + (i & 8) + i) * 128 + n]);
        oacc[in] = WMMA_BF16(af, bfv, oacc[in]);
      }
    }
    __syncthreads();
  }

  // ---- finalize: O / l, bf16 into cat[:, h*128 + ...] ----
  {
    const int h = bh & 15, b = bh >> 4;
    float linv[8];
    #pragma unroll
    for (int r = 0; r < 8; r++) linv[r] = 1.f / lrow[wm * 16 + mr8 + r];
    #pragma unroll
    for (int in = 0; in < 4; in++) {
      int col = h * 128 + wn * 64 + in * 16 + lane15;
      #pragma unroll
      for (int r = 0; r < 8; r++) {
        int row = qb0 + wm * 16 + mr8 + r;
        cat[(size_t)(b * 2048 + row) * 3072 + col] = f2us(oacc[in][r] * linv[r]);
      }
    }
  }
}

// ---------------- whiteboard: chunked GLA scan (chunk=64), state in LDS, WMMA everywhere ------
__global__ __launch_bounds__(256) void wb_kernel(const float* __restrict__ qwf,
                                                 const float* __restrict__ kwf,
                                                 const float* __restrict__ vwf,
                                                 const float* __restrict__ betaf,
                                                 const float* __restrict__ decayf,
                                                 const float* __restrict__ emb,
                                                 const float* __restrict__ st0,
                                                 float* __restrict__ wbo) {
  extern __shared__ char smem[];
  float* S0           = (float*)smem;                      // 128x256 fp32  (128 KB)
  unsigned short* Qs  = (unsigned short*)(S0 + 128 * 256); // 64x128 bf16
  unsigned short* Ksh = Qs + 64 * 128;                     // 64x128 bf16
  unsigned short* Vsh = Ksh + 64 * 128;                    // 64x256 bf16
  float* Sc           = (float*)(Vsh + 64 * 256);          // 64x64 fp32
  unsigned short* Ps  = (unsigned short*)(Sc + 64 * 64);   // 64x64 bf16
  float* Lc           = (float*)(Ps + 64 * 64);            // 64 cum log-decay
  float* Bv           = Lc + 64;                           // 64 beta
  float* Dl           = Bv + 64;                           // 64 log-decay

  const int tid = threadIdx.x, lane = tid & 31, wave = tid >> 5;
  const int wm = wave & 3, wn = wave >> 2;
  const int lane15 = lane & 15, k0l = (lane >> 4) * 8, kOff = (lane >> 4) * 16, mr8 = (lane >> 4) * 8;
  const int bh = blockIdx.x, h = bh & 3, b = bh >> 2;

  {   // load initial state [B,HW,128,256]
    const float* sg = st0 + (size_t)bh * 128 * 256;
    #pragma unroll
    for (int i = 0; i < 128; i++) S0[tid + i * 256] = sg[tid + i * 256];
  }
  __syncthreads();

  for (int c = 0; c < 32; c++) {
    const int s0 = c * 64;
    // ---- load chunk ----
    for (int i = 0; i < 32; i++) {
      int idx = tid + i * 256;         // 8192 = 64x128
      int rr = idx >> 7, d = idx & 127;
      size_t grow = (size_t)(b * 2048 + s0 + rr);
      Qs[rr * 128 + d]  = f2us(qwf[grow * 512 + h * 128 + d]);
      Ksh[rr * 128 + d] = f2us(kwf[grow * 512 + h * 128 + d] + emb[h * 128 + d]);
    }
    for (int i = 0; i < 64; i++) {
      int idx = tid + i * 256;         // 16384 = 64x256
      int rr = idx >> 8, d = idx & 255;
      size_t grow = (size_t)(b * 2048 + s0 + rr);
      Vsh[rr * 256 + d] = f2us(vwf[grow * 1024 + h * 256 + d]);
    }
    if (tid < 64) {
      size_t grow = (size_t)(b * 2048 + s0 + tid);
      Bv[tid] = betaf[grow * 4 + h];
      Dl[tid] = __logf(decayf[grow * 4 + h]);
    }
    __syncthreads();
    if (tid == 0) {
      float a = 0.f;
      for (int t = 0; t < 64; t++) { a += Dl[t]; Lc[t] = a; }
    }
    __syncthreads();

    // ---- QK^T -> Sc ----
    {
      f32x8 sacc[2];
      for (int j = 0; j < 2; j++) for (int e = 0; e < 8; e++) sacc[j][e] = 0.f;
      #pragma unroll
      for (int kk = 0; kk < 128; kk += 32) {
        bf16x16 af;
        int row = wm * 16 + lane15;
        #pragma unroll
        for (int i = 0; i < 8; i++) {
          af[i]     = us2bf(Qs[row * 128 + kk + k0l + i]);
          af[8 + i] = us2bf(Qs[row * 128 + kk + k0l + 16 + i]);
        }
        #pragma unroll
        for (int in = 0; in < 2; in++) {
          int n = wn * 32 + in * 16 + lane15;
          bf16x16 bfv;
          #pragma unroll
          for (int i = 0; i < 16; i++) bfv[i] = us2bf(Ksh[n * 128 + kk + k0l + (i & 8) + i]);
          sacc[in] = WMMA_BF16(af, bfv, sacc[in]);
        }
      }
      #pragma unroll
      for (int in = 0; in < 2; in++) {
        int col = wn * 32 + in * 16 + lane15;
        #pragma unroll
        for (int r = 0; r < 8; r++) Sc[(wm * 16 + mr8 + r) * 64 + col] = sacc[in][r];
      }
    }
    __syncthreads();

    // ---- decay-masked P ----
    {
      int row = tid >> 2, seg = tid & 3;
      float Lr = Lc[row];
      #pragma unroll
      for (int i = 0; i < 16; i++) {
        int col = seg * 16 + i;
        float p = (col <= row) ? Sc[row * 64 + col] * __expf(Lr - Lc[col]) * Bv[col] : 0.f;
        Ps[row * 64 + col] = f2us(p);
      }
    }
    __syncthreads();

    // ---- o = P@V + exp(L)*Q @ S0 ; wave -> rows 16*wm, cols 128*wn ----
    {
      f32x8 oacc[8];
      for (int j = 0; j < 8; j++) for (int e = 0; e < 8; e++) oacc[j][e] = 0.f;
      #pragma unroll
      for (int kk = 0; kk < 64; kk += 32) {
        bf16x16 af;
        int row = wm * 16 + lane15;
        #pragma unroll
        for (int i = 0; i < 8; i++) {
          af[i]     = us2bf(Ps[row * 64 + kk + k0l + i]);
          af[8 + i] = us2bf(Ps[row * 64 + kk + k0l + 16 + i]);
        }
        #pragma unroll
        for (int in = 0; in < 8; in++) {
          int n = wn * 128 + in * 16 + lane15;
          bf16x16 bfv;
          #pragma unroll
          for (int i = 0; i < 16; i++) bfv[i] = us2bf(Vsh[(kk + k0l + (i & 8) + i) * 256 + n]);
          oacc[in] = WMMA_BF16(af, bfv, oacc[in]);
        }
      }
      __syncthreads();
      // scale Q in place by exp(L_t)
      for (int i = 0; i < 32; i++) {
        int idx = tid + i * 256;
        int rr = idx >> 7, d = idx & 127;
        Qs[rr * 128 + d] = f2us(us2f(Qs[rr * 128 + d]) * __expf(Lc[rr]));
      }
      __syncthreads();
      #pragma unroll
      for (int kk = 0; kk < 128; kk += 32) {
        bf16x16 af;
        int row = wm * 16 + lane15;
        #pragma unroll
        for (int i = 0; i < 8; i++) {
          af[i]     = us2bf(Qs[row * 128 + kk + k0l + i]);
          af[8 + i] = us2bf(Qs[row * 128 + kk + k0l + 16 + i]);
        }
        #pragma unroll
        for (int in = 0; in < 8; in++) {
          int n = wn * 128 + in * 16 + lane15;
          bf16x16 bfv;
          #pragma unroll
          for (int i = 0; i < 16; i++) bfv[i] = f2bf(S0[(kk + k0l + (i & 8) + i) * 256 + n]);
          oacc[in] = WMMA_BF16(af, bfv, oacc[in]);
        }
      }
      #pragma unroll
      for (int in = 0; in < 8; in++) {
        int col = wn * 128 + in * 16 + lane15;
        #pragma unroll
        for (int r = 0; r < 8; r++) {
          size_t grow = (size_t)(b * 2048 + s0 + wm * 16 + mr8 + r);
          wbo[grow * 1024 + h * 256 + col] = oacc[in][r];
        }
      }
    }
    __syncthreads();

    // ---- K' = exp(L63 - L_t) * beta_t * k_t ----
    {
      float L63 = Lc[63];
      for (int i = 0; i < 32; i++) {
        int idx = tid + i * 256;
        int rr = idx >> 7, d = idx & 127;
        Ksh[rr * 128 + d] = f2us(us2f(Ksh[rr * 128 + d]) * __expf(L63 - Lc[rr]) * Bv[rr]);
      }
    }
    __syncthreads();

    // ---- S0 = exp(L63)*S0 + K'^T @ V ; wave -> rows 16*wave, all 256 cols ----
    {
      float dAll = __expf(Lc[63]);
      f32x8 up[16];
      for (int j = 0; j < 16; j++) for (int e = 0; e < 8; e++) up[j][e] = 0.f;
      #pragma unroll
      for (int kk = 0; kk < 64; kk += 32) {
        bf16x16 af;
        int m = wave * 16 + lane15;
        #pragma unroll
        for (int i = 0; i < 8; i++) {
          af[i]     = us2bf(Ksh[(kk + k0l + i) * 128 + m]);
          af[8 + i] = us2bf(Ksh[(kk + k0l + 16 + i) * 128 + m]);
        }
        #pragma unroll
        for (int in = 0; in < 16; in++) {
          int n = in * 16 + lane15;
          bf16x16 bfv;
          #pragma unroll
          for (int i = 0; i < 16; i++) bfv[i] = us2bf(Vsh[(kk + k0l + (i & 8) + i) * 256 + n]);
          up[in] = WMMA_BF16(af, bfv, up[in]);
        }
      }
      #pragma unroll
      for (int in = 0; in < 16; in++) {
        int n = in * 16 + lane15;
        #pragma unroll
        for (int r = 0; r < 8; r++) {
          int m = wave * 16 + mr8 + r;
          S0[m * 256 + n] = dAll * S0[m * 256 + n] + up[in][r];
        }
      }
    }
    __syncthreads();
  }
}

// ---------------- gated RMSNorm, writes bf16 into cat[:, 2048:3072] ----------------
__global__ __launch_bounds__(256) void norm_kernel(const float* __restrict__ wbo,
                                                   const float* __restrict__ gatef,
                                                   const float* __restrict__ normw,
                                                   unsigned short* __restrict__ cat) {
  __shared__ float red[8];
  int row = blockIdx.x, tid = threadIdx.x, lane = tid & 31, wave = tid >> 5;
  for (int h = 0; h < 4; h++) {
    float v = wbo[(size_t)row * 1024 + h * 256 + tid];
    float ss = v * v;
    for (int m = 1; m < 32; m <<= 1) ss += __shfl_xor(ss, m, 32);
    if (lane == 0) red[wave] = ss;
    __syncthreads();
    float tot = 0.f;
    for (int i = 0; i < 8; i++) tot += red[i];
    float inv = rsqrtf(tot * (1.f / 256.f) + 1e-5f);
    float g = gatef[(size_t)row * 1024 + h * 256 + tid];
    float o = v * inv * normw[tid] * (1.f / (1.f + __expf(-g)));
    cat[(size_t)row * 3072 + 2048 + h * 256 + tid] = f2us(o);
    __syncthreads();
  }
}

// ==================== launch ====================
extern "C" void kernel_launch(void* const* d_in, const int* in_sizes, int n_in,
                              void* d_out, int out_size, void* d_ws, size_t ws_size,
                              hipStream_t stream) {
  (void)in_sizes; (void)n_in; (void)out_size; (void)ws_size;
  const float* hs     = (const float*)d_in[0];
  const float* st0    = (const float*)d_in[1];
  const float* Wq     = (const float*)d_in[2];
  const float* Wk     = (const float*)d_in[3];
  const float* Wv     = (const float*)d_in[4];
  const float* Wwbq   = (const float*)d_in[5];
  const float* Wwbk   = (const float*)d_in[6];
  const float* Wwbv   = (const float*)d_in[7];
  const float* Wbeta  = (const float*)d_in[8];
  const float* Wdecay = (const float*)d_in[9];
  const float* bdec   = (const float*)d_in[10];
  const float* emb    = (const float*)d_in[11];
  const float* Wg     = (const float*)d_in[12];
  const float* normw  = (const float*)d_in[13];
  const float* Wo     = (const float*)d_in[14];
  float* out = (float*)d_out;

  char* ws = (char*)d_ws;
  size_t off = 0;
  auto carve = [&](size_t bytes) -> char* {
    char* p = ws + off; off += (bytes + 255) & ~(size_t)255; return p;
  };
  unsigned short* hbf  = (unsigned short*)carve(4096ull * 2048 * 2);
  unsigned short* wbuf = (unsigned short*)carve(3072ull * 2048 * 2);   // staging for bf16 weights
  float* qf    = (float*)carve(4096ull * 2048 * 4);
  float* kf    = (float*)carve(4096ull * 2048 * 4);
  float* vf    = (float*)carve(4096ull * 2048 * 4);
  unsigned short* qr = (unsigned short*)carve(4096ull * 2048 * 2);
  unsigned short* kr = (unsigned short*)carve(4096ull * 2048 * 2);
  unsigned short* vr = (unsigned short*)carve(4096ull * 2048 * 2);
  float* qwf   = (float*)carve(4096ull * 512 * 4);
  float* kwf   = (float*)carve(4096ull * 512 * 4);
  float* vwf   = (float*)carve(4096ull * 1024 * 4);
  float* gatef = (float*)carve(4096ull * 1024 * 4);
  float* betaf = (float*)carve(4096ull * 4 * 4);
  float* decayf= (float*)carve(4096ull * 4 * 4);
  // aliases: qf/kf are dead after rope_kernel; reuse their memory
  unsigned short* cat = (unsigned short*)qf;   // 4096x3072 bf16 (24 MB <= 32 MB)
  float* wbo = kf;                             // 4096x1024 fp32 (16 MB <= 32 MB)

  const dim3 blk(256);
  const size_t gemm_lds = 16384;

  // hidden states -> bf16
  cvt_bf16_kernel<<<(4096 * 2048) / 256, blk, 0, stream>>>(hs, hbf, 4096 * 2048);

  // attention projections
  cvt_bf16_kernel<<<(2048 * 2048) / 256, blk, 0, stream>>>(Wq, wbuf, 2048 * 2048);
  gemm_bf16_kernel<<<dim3(16, 32), blk, gemm_lds, stream>>>(hbf, wbuf, qf, 4096, 2048, 2048);
  cvt_bf16_kernel<<<(2048 * 2048) / 256, blk, 0, stream>>>(Wk, wbuf, 2048 * 2048);
  gemm_bf16_kernel<<<dim3(16, 32), blk, gemm_lds, stream>>>(hbf, wbuf, kf, 4096, 2048, 2048);
  cvt_bf16_kernel<<<(2048 * 2048) / 256, blk, 0, stream>>>(Wv, wbuf, 2048 * 2048);
  gemm_bf16_kernel<<<dim3(16, 32), blk, gemm_lds, stream>>>(hbf, wbuf, vf, 4096, 2048, 2048);

  // whiteboard projections + gate
  cvt_bf16_kernel<<<(2048 * 512) / 256, blk, 0, stream>>>(Wwbq, wbuf, 2048 * 512);
  gemm_bf16_kernel<<<dim3(4, 32), blk, gemm_lds, stream>>>(hbf, wbuf, qwf, 4096, 512, 2048);
  cvt_bf16_kernel<<<(2048 * 512) / 256, blk, 0, stream>>>(Wwbk, wbuf, 2048 * 512);
  gemm_bf16_kernel<<<dim3(4, 32), blk, gemm_lds, stream>>>(hbf, wbuf, kwf, 4096, 512, 2048);
  cvt_bf16_kernel<<<(2048 * 1024) / 256, blk, 0, stream>>>(Wwbv, wbuf, 2048 * 1024);
  gemm_bf16_kernel<<<dim3(8, 32), blk, gemm_lds, stream>>>(hbf, wbuf, vwf, 4096, 1024, 2048);
  cvt_bf16_kernel<<<(2048 * 1024) / 256, blk, 0, stream>>>(Wg, wbuf, 2048 * 1024);
  gemm_bf16_kernel<<<dim3(8, 32), blk, gemm_lds, stream>>>(hbf, wbuf, gatef, 4096, 1024, 2048);

  // beta / decay (N=4: VALU reduction kernel)
  beta_decay_kernel<<<4096, blk, 0, stream>>>(hs, Wbeta, Wdecay, bdec, betaf, decayf);

  // RoPE + relayout (consumes qf/kf/vf)
  rope_kernel<<<dim3(4096, 16), dim3(128), 0, stream>>>(qf, kf, vf, qr, kr, vr);

  // flash attention -> cat[:, 0:2048]
  {
    size_t shmem = (size_t)(3 * 64 * 128 * 2) + 64 * 64 * 4 + 64 * 64 * 2 + 3 * 64 * 4;
    attn_kernel<<<dim3(32, 32), blk, shmem, stream>>>(qr, kr, vr, cat);
  }

  // whiteboard chunked scan -> wbo
  {
    size_t shmem = (size_t)(128 * 256 * 4) + 2 * (64 * 128 * 2) + 64 * 256 * 2 +
                   64 * 64 * 4 + 64 * 64 * 2 + 3 * 64 * 4;
    wb_kernel<<<8, blk, shmem, stream>>>(qwf, kwf, vwf, betaf, decayf, emb, st0, wbo);
  }

  // gated RMSNorm -> cat[:, 2048:3072]
  norm_kernel<<<4096, blk, 0, stream>>>(wbo, gatef, normw, cat);

  // final projection: cat[4096x3072] @ Wo[3072x2048] -> out
  cvt_bf16_kernel<<<(3072 * 2048) / 256, blk, 0, stream>>>(Wo, wbuf, 3072 * 2048);
  gemm_bf16_kernel<<<dim3(16, 32), blk, gemm_lds, stream>>>(cat, wbuf, out, 4096, 2048, 3072);
}
